// MultiheadAttention_54726473286036
// MI455X (gfx1250) — matrix operand order repaired and checked by reading, not compiled, not run on hardware
//
#include <hip/hip_runtime.h>
#include <hip/hip_bf16.h>
#include <stdint.h>

typedef __bf16 bf16;
typedef __attribute__((ext_vector_type(16))) __bf16 v16bf;
typedef __attribute__((ext_vector_type(8)))  float  v8f;

#define L_SEQ 2048
#define BATCH 2
#define EMB   768
#define HEADS 12
#define HD    64
#define RANK  64

// round-to-nearest-even float -> bf16 without relying on __bf16 cvt codegen
__device__ __forceinline__ bf16 tobf(float f) {
  unsigned u = __float_as_uint(f);
  unsigned r = (u + 0x7FFFu + ((u >> 16) & 1u)) >> 16;
  unsigned short s = (unsigned short)r;
  return __builtin_bit_cast(bf16, s);
}

// A-fragment K index map for 16-bit 16x32 A matrix: contiguous k = 16*half + e (matches B fills)
__device__ __forceinline__ int kmapA(int e, int half) {
  return e + 16 * half;
}

// cross-lane xor swap within 16-lane group via ds_swizzle (wave32, EXEC all-ones)
#define SWIZ_F(x, imm) __int_as_float(__builtin_amdgcn_ds_swizzle(__float_as_int(x), (imm)))

// CDNA5 async global->LDS copy, 16B per lane (ASYNCcnt-tracked).
// LDS operand = low 32 bits of the generic pointer (ISA 10.2).
__device__ __forceinline__ void async_ld_b128(const bf16* g, bf16* l) {
  unsigned loff = (unsigned)(unsigned long long)(size_t)l;
  unsigned long long ga = (unsigned long long)(size_t)g;
  asm volatile("global_load_async_to_lds_b128 %0, %1, off"
               :: "v"(loff), "v"(ga) : "memory");
}
__device__ __forceinline__ void async_wait0() {
  asm volatile("s_wait_asynccnt 0x0" ::: "memory");
}

__global__ void f2bf_kernel(const float* __restrict__ in, bf16* __restrict__ out, int n) {
  for (int i = blockIdx.x * blockDim.x + threadIdx.x; i < n; i += gridDim.x * blockDim.x)
    out[i] = tobf(in[i]);
}

__global__ void zerof_kernel(float* __restrict__ p, int n) {
  for (int i = blockIdx.x * blockDim.x + threadIdx.x; i < n; i += gridDim.x * blockDim.x)
    p[i] = 0.0f;
}

// ---------------------------------------------------------------------------
// bf16 WMMA GEMM: C(MxN) = A(MxK) @ W(NxK)^T, fp32 accum.
// WG tile 128x64, 8 waves (4x2), each wave 32x32 = 2x2 WMMA tiles per K-step
// (4 wmma per 8 ds_load_b128). Global->LDS staging via async-to-LDS.
// MODE 0: relu(acc+bias) -> bf16 | MODE 1: acc+bias -> bf16
// MODE 2: SSF + fused LoRA-up, split-store q(/8)/k/v per-head bf16
// MODE 3: SSF + fused LoRA-up -> fp32
// ---------------------------------------------------------------------------
template <int MODE>
__global__ __launch_bounds__(256) void gemm_kernel(
    const bf16* __restrict__ A, const bf16* __restrict__ W,
    const float* __restrict__ bias, const float* __restrict__ scale,
    const float* __restrict__ shift, const bf16* __restrict__ A2,
    const bf16* __restrict__ W2, const float* __restrict__ bias2,
    const float* __restrict__ lscale, int N, int K, int K2,
    bf16* __restrict__ outb, bf16* __restrict__ outq, bf16* __restrict__ outk,
    bf16* __restrict__ outv, float* __restrict__ outf) {
  __shared__ bf16 As[128][40];   // +8 pad, rows stay 16B aligned
  __shared__ bf16 Ws[64][40];

  const int tid  = threadIdx.x;
  const int wave = tid >> 5, lane = tid & 31;
  const int wm = wave & 3, wn = wave >> 2;         // 4 x 2 wave grid
  const int half = lane >> 4, l16 = lane & 15;
  const int row0 = blockIdx.x * 128, col0 = blockIdx.y * 64;

  v8f acc[2][2] = {};
  for (int k0 = 0; k0 < K; k0 += 32) {
    __syncthreads();
    {
      int r = tid >> 1, cv = (tid & 1) * 16;       // A: 128 rows x 32 cols
      const bf16* g = A + (size_t)(row0 + r) * K + k0 + cv;
      async_ld_b128(g, &As[r][cv]);
      async_ld_b128(g + 8, &As[r][cv + 8]);
    }
    {
      int r = tid >> 2, cv = (tid & 3) * 8;        // W: 64 rows x 32 cols
      async_ld_b128(W + (size_t)(col0 + r) * K + k0 + cv, &Ws[r][cv]);
    }
    async_wait0();
    __syncthreads();

    v16bf a0, a1, b0, b1;
#pragma unroll
    for (int e = 0; e < 16; ++e) {
      int km = kmapA(e, half);
      a0[e] = As[wm * 32 + l16][km];
      a1[e] = As[wm * 32 + 16 + l16][km];
    }
#pragma unroll
    for (int e = 0; e < 16; ++e) {
      b0[e] = Ws[wn * 32 + l16][e + 16 * half];
      b1[e] = Ws[wn * 32 + 16 + l16][e + 16 * half];
    }
    acc[0][0] = __builtin_amdgcn_wmma_f32_16x16x32_bf16(false, a0, false, b0, (short)0, acc[0][0], false, false);
    acc[0][1] = __builtin_amdgcn_wmma_f32_16x16x32_bf16(false, a0, false, b1, (short)0, acc[0][1], false, false);
    acc[1][0] = __builtin_amdgcn_wmma_f32_16x16x32_bf16(false, a1, false, b0, (short)0, acc[1][0], false, false);
    acc[1][1] = __builtin_amdgcn_wmma_f32_16x16x32_bf16(false, a1, false, b1, (short)0, acc[1][1], false, false);
  }

  v8f acc2[2][2] = {};
  if (MODE >= 2) {  // fused LoRA-up second GEMM (K2 = 64)
    for (int k0 = 0; k0 < K2; k0 += 32) {
      __syncthreads();
      {
        int r = tid >> 1, cv = (tid & 1) * 16;
        const bf16* g = A2 + (size_t)(row0 + r) * K2 + k0 + cv;
        async_ld_b128(g, &As[r][cv]);
        async_ld_b128(g + 8, &As[r][cv + 8]);
      }
      {
        int r = tid >> 2, cv = (tid & 3) * 8;
        async_ld_b128(W2 + (size_t)(col0 + r) * K2 + k0 + cv, &Ws[r][cv]);
      }
      async_wait0();
      __syncthreads();

      v16bf a0, a1, b0, b1;
#pragma unroll
      for (int e = 0; e < 16; ++e) {
        int km = kmapA(e, half);
        a0[e] = As[wm * 32 + l16][km];
        a1[e] = As[wm * 32 + 16 + l16][km];
      }
#pragma unroll
      for (int e = 0; e < 16; ++e) {
        b0[e] = Ws[wn * 32 + l16][e + 16 * half];
        b1[e] = Ws[wn * 32 + 16 + l16][e + 16 * half];
      }
      acc2[0][0] = __builtin_amdgcn_wmma_f32_16x16x32_bf16(false, a0, false, b0, (short)0, acc2[0][0], false, false);
      acc2[0][1] = __builtin_amdgcn_wmma_f32_16x16x32_bf16(false, a0, false, b1, (short)0, acc2[0][1], false, false);
      acc2[1][0] = __builtin_amdgcn_wmma_f32_16x16x32_bf16(false, a1, false, b0, (short)0, acc2[1][0], false, false);
      acc2[1][1] = __builtin_amdgcn_wmma_f32_16x16x32_bf16(false, a1, false, b1, (short)0, acc2[1][1], false, false);
    }
  }

#pragma unroll
  for (int mt = 0; mt < 2; ++mt) {
#pragma unroll
    for (int nt = 0; nt < 2; ++nt) {
      const int c = col0 + wn * 32 + nt * 16 + l16;
      const int rbase = row0 + wm * 32 + mt * 16;
      if (MODE == 0) {
        float bc = bias[c];
#pragma unroll
        for (int v = 0; v < 8; ++v) {
          float x = acc[mt][nt][v] + bc;
          x = x > 0.0f ? x : 0.0f;
          outb[(size_t)(rbase + v + 8 * half) * N + c] = tobf(x);
        }
      } else if (MODE == 1) {
        float bc = bias[c];
#pragma unroll
        for (int v = 0; v < 8; ++v)
          outb[(size_t)(rbase + v + 8 * half) * N + c] = tobf(acc[mt][nt][v] + bc);
      } else if (MODE == 2) {
        float bc = bias[c], s1 = scale[c], sh = shift[c], b2 = bias2[c], ls = lscale[0];
        int part = c / EMB;
        int e = c - part * EMB;
        int hh = e >> 6, d = e & 63;
        bf16* dst = (part == 0) ? outq : ((part == 1) ? outk : outv);
        float mul = (part == 0) ? 0.125f : 1.0f;   // hd^-0.5
#pragma unroll
        for (int v = 0; v < 8; ++v) {
          int r = rbase + v + 8 * half;             // r = i*B + b
          int i = r >> 1, bidx = r & 1;
          float x = ((acc[mt][nt][v] + bc) * s1 + sh + (acc2[mt][nt][v] + b2) * ls) * mul;
          dst[(((size_t)(bidx * HEADS + hh)) * L_SEQ + i) * HD + d] = tobf(x);
        }
      } else {
        float bc = bias[c], s1 = scale[c], sh = shift[c], b2 = bias2[c], ls = lscale[0];
#pragma unroll
        for (int v = 0; v < 8; ++v)
          outf[(size_t)(rbase + v + 8 * half) * N + c] =
              (acc[mt][nt][v] + bc) * s1 + sh + (acc2[mt][nt][v] + b2) * ls;
      }
    }
  }
}

// ---------------------------------------------------------------------------
// Attention: grid (L/128, B*H). 8 waves/WG, each wave owns 16 q rows.
// Pass 1: exact row max+sum (WMMA S tiles + ds_swizzle butterflies).
// Pass 2: normalized P -> atomicAdd head-mean, O = P@V via WMMA.
// K/V staged with async global->LDS.
// ---------------------------------------------------------------------------
__global__ __launch_bounds__(256) void attn_kernel(
    const bf16* __restrict__ qh, const bf16* __restrict__ kh,
    const bf16* __restrict__ vh, bf16* __restrict__ attn_o,
    float* __restrict__ aw) {
  __shared__ bf16 Ks[64][72];
  __shared__ bf16 Vs[64][72];
  __shared__ bf16 Ps[8][16][72];

  const int tid  = threadIdx.x;
  const int wave = tid >> 5, lane = tid & 31;
  const int half = lane >> 4, l16 = lane & 15;
  const int bh = blockIdx.y;
  const int bidx = bh / HEADS, hh = bh % HEADS;
  const int i0 = blockIdx.x * 128 + wave * 16;

  const bf16* qb = qh + (size_t)bh * L_SEQ * HD;
  const bf16* kb = kh + (size_t)bh * L_SEQ * HD;
  const bf16* vb = vh + (size_t)bh * L_SEQ * HD;

  v16bf qf[2];
#pragma unroll
  for (int kk = 0; kk < 2; ++kk)
#pragma unroll
    for (int e = 0; e < 16; ++e)
      qf[kk][e] = qb[(size_t)(i0 + l16) * HD + kk * 32 + kmapA(e, half)];

  float m[8], l[8];
#pragma unroll
  for (int v = 0; v < 8; ++v) { m[v] = -1e30f; l[v] = 0.0f; }

  // ---- pass 1 ----------------------------------------------------------
  for (int j0 = 0; j0 < L_SEQ; j0 += 64) {
    __syncthreads();
    {
      int r = tid >> 2, cv = (tid & 3) * 16;
      const bf16* g = kb + (size_t)(j0 + r) * HD + cv;
      async_ld_b128(g, &Ks[r][cv]);
      async_ld_b128(g + 8, &Ks[r][cv + 8]);
    }
    async_wait0();
    __syncthreads();
#pragma unroll
    for (int jt = 0; jt < 4; ++jt) {
      v8f s = {};
#pragma unroll
      for (int kk = 0; kk < 2; ++kk) {
        v16bf bfr;
#pragma unroll
        for (int e = 0; e < 16; ++e)
          bfr[e] = Ks[jt * 16 + l16][kk * 32 + e + 16 * half];
        s = __builtin_amdgcn_wmma_f32_16x16x32_bf16(false, qf[kk], false, bfr,
                                                    (short)0, s, false, false);
      }
#pragma unroll
      for (int v = 0; v < 8; ++v) {
        float rm = s[v];
        rm = fmaxf(rm, SWIZ_F(rm, 0x041f));
        rm = fmaxf(rm, SWIZ_F(rm, 0x081f));
        rm = fmaxf(rm, SWIZ_F(rm, 0x101f));
        rm = fmaxf(rm, SWIZ_F(rm, 0x201f));
        float mn = fmaxf(m[v], rm);
        float p = __expf(s[v] - mn);
        float rs = p;
        rs += SWIZ_F(rs, 0x041f);
        rs += SWIZ_F(rs, 0x081f);
        rs += SWIZ_F(rs, 0x101f);
        rs += SWIZ_F(rs, 0x201f);
        l[v] = l[v] * __expf(m[v] - mn) + rs;
        m[v] = mn;
      }
    }
  }
  float rl[8];
#pragma unroll
  for (int v = 0; v < 8; ++v) rl[v] = 1.0f / l[v];

  // ---- pass 2 ----------------------------------------------------------
  v8f oacc[4] = {};
  for (int j0 = 0; j0 < L_SEQ; j0 += 64) {
    __syncthreads();
    {
      int r = tid >> 2, cv = (tid & 3) * 16;
      const bf16* gk = kb + (size_t)(j0 + r) * HD + cv;
      async_ld_b128(gk, &Ks[r][cv]);
      async_ld_b128(gk + 8, &Ks[r][cv + 8]);
      const bf16* gv = vb + (size_t)(j0 + r) * HD + cv;
      async_ld_b128(gv, &Vs[r][cv]);
      async_ld_b128(gv + 8, &Vs[r][cv + 8]);
    }
    async_wait0();
    __syncthreads();
#pragma unroll
    for (int jt = 0; jt < 4; ++jt) {
      v8f s = {};
#pragma unroll
      for (int kk = 0; kk < 2; ++kk) {
        v16bf bfr;
#pragma unroll
        for (int e = 0; e < 16; ++e)
          bfr[e] = Ks[jt * 16 + l16][kk * 32 + e + 16 * half];
        s = __builtin_amdgcn_wmma_f32_16x16x32_bf16(false, qf[kk], false, bfr,
                                                    (short)0, s, false, false);
      }
      int jcol = j0 + jt * 16 + l16;
#pragma unroll
      for (int v = 0; v < 8; ++v) {
        float p = __expf(s[v] - m[v]) * rl[v];
        int irow = i0 + v + 8 * half;
        atomicAdd(&aw[((size_t)bidx * L_SEQ + irow) * L_SEQ + jcol],
                  p * (1.0f / (float)HEADS));
        Ps[wave][v + 8 * half][jt * 16 + l16] = tobf(p);
      }
    }
    __syncthreads();
#pragma unroll
    for (int kk2 = 0; kk2 < 2; ++kk2) {
      v16bf pf;
#pragma unroll
      for (int e = 0; e < 16; ++e)
        pf[e] = Ps[wave][l16][kk2 * 32 + kmapA(e, half)];
#pragma unroll
      for (int dt = 0; dt < 4; ++dt) {
        v16bf vf;
#pragma unroll
        for (int e = 0; e < 16; ++e)
          vf[e] = Vs[kk2 * 32 + e + 16 * half][dt * 16 + l16];
        oacc[dt] = __builtin_amdgcn_wmma_f32_16x16x32_bf16(false, pf, false, vf,
                                                           (short)0, oacc[dt],
                                                           false, false);
      }
    }
  }

#pragma unroll
  for (int dt = 0; dt < 4; ++dt)
#pragma unroll
    for (int v = 0; v < 8; ++v) {
      int irow = i0 + v + 8 * half;
      attn_o[((size_t)irow * BATCH + bidx) * EMB + hh * HD + dt * 16 + l16] =
          tobf(oacc[dt][v]);
    }
}

// ---------------------------------------------------------------------------
extern "C" void kernel_launch(void* const* d_in, const int* in_sizes, int n_in,
                              void* d_out, int out_size, void* d_ws, size_t ws_size,
                              hipStream_t stream) {
  (void)in_sizes; (void)n_in; (void)out_size; (void)ws_size;
  const float* query     = (const float*)d_in[0];
  const float* in_proj_w = (const float*)d_in[3];
  const float* in_proj_b = (const float*)d_in[4];
  const float* ssf_s1    = (const float*)d_in[5];
  const float* ssf_sh1   = (const float*)d_in[6];
  const float* ssf_s2    = (const float*)d_in[7];
  const float* ssf_sh2   = (const float*)d_in[8];
  const float* out_w     = (const float*)d_in[9];
  const float* out_b     = (const float*)d_in[10];
  const float* l1_dw     = (const float*)d_in[11];
  const float* l1_db     = (const float*)d_in[12];
  const float* l1_uw     = (const float*)d_in[13];
  const float* l1_ub     = (const float*)d_in[14];
  const float* l1_s      = (const float*)d_in[15];
  const float* l2_dw     = (const float*)d_in[16];
  const float* l2_db     = (const float*)d_in[17];
  const float* l2_uw     = (const float*)d_in[18];
  const float* l2_ub     = (const float*)d_in[19];
  const float* l2_s      = (const float*)d_in[20];

  const int MB = L_SEQ * BATCH;            // 4096 rows
  char* w = (char*)d_ws;
  size_t off = 0;
  auto alloc = [&](size_t nelem) -> bf16* {
    bf16* p = (bf16*)(w + off);
    off = (off + nelem * sizeof(bf16) + 255) & ~(size_t)255;
    return p;
  };
  bf16* q_bf   = alloc((size_t)MB * EMB);
  bf16* w_in   = alloc((size_t)3 * EMB * EMB);
  bf16* w_out  = alloc((size_t)EMB * EMB);
  bf16* l1dw_b = alloc((size_t)RANK * EMB);
  bf16* l1uw_b = alloc((size_t)3 * EMB * RANK);
  bf16* l2dw_b = alloc((size_t)RANK * EMB);
  bf16* l2uw_b = alloc((size_t)EMB * RANK);
  bf16* mid1   = alloc((size_t)MB * RANK);
  bf16* mid2   = alloc((size_t)MB * RANK);
  bf16* qhb    = alloc((size_t)BATCH * HEADS * L_SEQ * HD);
  bf16* khb    = alloc((size_t)BATCH * HEADS * L_SEQ * HD);
  bf16* vhb    = alloc((size_t)BATCH * HEADS * L_SEQ * HD);
  bf16* attn_o = alloc((size_t)MB * EMB);
  bf16* out1   = alloc((size_t)MB * EMB);

  float* out2 = (float*)d_out;                    // (L,B,E)
  float* aw   = out2 + (size_t)MB * EMB;          // (B,L,L)

  auto cvt = [&](const float* src, bf16* dst, int n) {
    int blocks = (n + 256 * 8 - 1) / (256 * 8);
    f2bf_kernel<<<blocks, 256, 0, stream>>>(src, dst, n);
  };
  cvt(query, q_bf, MB * EMB);
  cvt(in_proj_w, w_in, 3 * EMB * EMB);
  cvt(out_w, w_out, EMB * EMB);
  cvt(l1_dw, l1dw_b, RANK * EMB);
  cvt(l1_uw, l1uw_b, 3 * EMB * RANK);
  cvt(l2_dw, l2dw_b, RANK * EMB);
  cvt(l2_uw, l2uw_b, EMB * RANK);

  // LoRA1 down: mid1 = relu(query @ l1_dw^T + l1_db)
  gemm_kernel<0><<<dim3(MB / 128, RANK / 64), 256, 0, stream>>>(
      q_bf, l1dw_b, l1_db, nullptr, nullptr, nullptr, nullptr, nullptr, nullptr,
      RANK, EMB, 0, mid1, nullptr, nullptr, nullptr, nullptr);

  // qkv = SSF(query@Win^T + b) + (mid1@l1_uw^T + l1_ub)*l1_s, split q/k/v
  gemm_kernel<2><<<dim3(MB / 128, 3 * EMB / 64), 256, 0, stream>>>(
      q_bf, w_in, in_proj_b, ssf_s1, ssf_sh1, mid1, l1uw_b, l1_ub, l1_s,
      3 * EMB, EMB, RANK, nullptr, qhb, khb, vhb, nullptr);

  // zero attn_weights accumulation target
  {
    int n = BATCH * L_SEQ * L_SEQ;
    zerof_kernel<<<(n + 256 * 16 - 1) / (256 * 16), 256, 0, stream>>>(aw, n);
  }

  attn_kernel<<<dim3(L_SEQ / 128, BATCH * HEADS), 256, 0, stream>>>(
      qhb, khb, vhb, attn_o, aw);

  // out1 = attn @ out_w^T + out_b
  gemm_kernel<1><<<dim3(MB / 128, EMB / 64), 256, 0, stream>>>(
      attn_o, w_out, out_b, nullptr, nullptr, nullptr, nullptr, nullptr, nullptr,
      EMB, EMB, 0, out1, nullptr, nullptr, nullptr, nullptr);

  // LoRA2 down: mid2 = relu(out1 @ l2_dw^T + l2_db)
  gemm_kernel<0><<<dim3(MB / 128, RANK / 64), 256, 0, stream>>>(
      out1, l2dw_b, l2_db, nullptr, nullptr, nullptr, nullptr, nullptr, nullptr,
      RANK, EMB, 0, mid2, nullptr, nullptr, nullptr, nullptr);

  // out2 = SSF2(out1 @ out_w^T + out_b) + (mid2@l2_uw^T + l2_ub)*l2_s
  gemm_kernel<3><<<dim3(MB / 128, EMB / 64), 256, 0, stream>>>(
      out1, w_out, out_b, ssf_s2, ssf_sh2, mid2, l2uw_b, l2_ub, l2_s,
      EMB, EMB, RANK, nullptr, nullptr, nullptr, nullptr, out2);
}
